// GRUCell_53970559041913
// MI455X (gfx1250) — matrix operand order repaired and checked by reading, not compiled, not run on hardware
//
#include <hip/hip_runtime.h>
#include <stdint.h>

// Problem constants (from reference): B=64, T=512, D=1024, H=1024
constexpr int BATCH = 64;
constexpr int SEQ   = 512;
constexpr int DIN   = 1024;
constexpr int HID   = 1024;
constexpr int KTOT  = 2048;            // D + H
constexpr int KSPLIT = 8;              // split-K factor
constexpr int KSPAN  = KTOT / KSPLIT;  // 256 K per block
constexpr int NCHUNK = KSPAN / 64;     // 4 LDS stages of K=64
constexpr int LDK    = 80;             // padded LDS row (bf16 elems), 160B = 16B aligned

typedef unsigned short u16;
typedef __attribute__((ext_vector_type(16))) __bf16 v16bf;
typedef __attribute__((ext_vector_type(8)))  float  v8f;
typedef int v4i_ __attribute__((vector_size(16)));   // 128-bit payload type for async LDS builtin

union FragBF { v16bf v; uint4 q[2]; };

__device__ __forceinline__ u16 f2bf(float f) {
  union { float f; uint32_t u; } c; c.f = f;
  uint32_t r = c.u + 0x7FFFu + ((c.u >> 16) & 1u);   // round-to-nearest-even
  return (u16)(r >> 16);
}

// ---------------------------------------------------------------------------
// CDNA5 async global->LDS path (GLOBAL_LOAD_ASYNC_TO_LDS_B128, ASYNCcnt).
// Probe-derived signature on this toolchain: (v4i* gaddr, v4i* ldsaddr, Ii, Ii)
// with generic pointers. Guarded: falls back to global_load_b128 +
// ds_store_b128 if the builtins are not declared.
// ---------------------------------------------------------------------------
#if __has_builtin(__builtin_amdgcn_global_load_async_to_lds_b128) && \
    __has_builtin(__builtin_amdgcn_s_wait_asynccnt)
#define USE_ASYNC_LDS 1
#else
#define USE_ASYNC_LDS 0
#endif

#if USE_ASYNC_LDS
__device__ __forceinline__ void cp_async_b128(const void* gsrc, void* ldst) {
  __builtin_amdgcn_global_load_async_to_lds_b128((v4i_*)gsrc, (v4i_*)ldst,
                                                 /*imm offset*/0, /*cpol*/0);
}
#endif

__device__ __forceinline__ void stage_fence() {
#if USE_ASYNC_LDS
  __builtin_amdgcn_s_wait_asynccnt(0);   // async LDS writes use ASYNCcnt, not DScnt
#endif
  __syncthreads();
}

// ---------------------------------------------------------------------------
// Split-K bf16 WMMA GEMM: C_part[ks] += A(64xKTOT) @ Bt^T, Bt stored [NOUT][KTOT]
// A is the concat [A0 | A1] along K (each [64][1024], K-contiguous bf16).
// Block: 256 threads = 8 waves (wave32). Tile: M=64, N=128, Kspan=256.
// Each wave: 32x32 = 4x v_wmma_f32_16x16x32_bf16 accumulators.
// ---------------------------------------------------------------------------
template <int NOUT>
__global__ void __launch_bounds__(256) gru_gemm_bf16(
    const u16* __restrict__ A0, const u16* __restrict__ A1,
    const u16* __restrict__ Bt, float* __restrict__ out_part)
{
  __shared__ __align__(16) u16 As[2][BATCH][LDK];  // 20.5 KB
  __shared__ __align__(16) u16 Bs[2][128][LDK];    // 41 KB

  const int tid = threadIdx.x;
  const int n0  = blockIdx.x * 128;
  const int k0  = blockIdx.y * KSPAN;

  auto copy_stage = [&](int buf, int kc) {
    // As tile: 64 rows x 64 bf16 = 512 x b128; 2 per thread
#pragma unroll
    for (int r = 0; r < 2; ++r) {
      int i   = tid + r * 256;
      int row = i >> 3;
      int col = (i & 7) * 8;
      int kk  = kc + col;
      const u16* src = (kk < DIN) ? (A0 + row * DIN + kk)
                                  : (A1 + row * HID + (kk - DIN));
#if USE_ASYNC_LDS
      cp_async_b128(src, &As[buf][row][col]);
#else
      *(uint4*)&As[buf][row][col] = *(const uint4*)src;
#endif
    }
    // Bs tile: 128 rows x 64 bf16 = 1024 x b128; 4 per thread
#pragma unroll
    for (int r = 0; r < 4; ++r) {
      int i   = tid + r * 256;
      int row = i >> 3;
      int col = (i & 7) * 8;
      const u16* src = Bt + (size_t)(n0 + row) * KTOT + kc + col;
#if USE_ASYNC_LDS
      cp_async_b128(src, &Bs[buf][row][col]);
#else
      *(uint4*)&Bs[buf][row][col] = *(const uint4*)src;
#endif
    }
  };

  const int lane = tid & 31;
  const int w    = tid >> 5;
  const int wm   = w >> 2;        // 0..1 -> M offset 32*wm
  const int wn   = w & 3;         // 0..3 -> N offset 32*wn
  const int l    = lane & 15;
  const int hi   = lane >> 4;     // lane half selects K octets (ISA layout)

  v8f acc[2][2] = {};

  copy_stage(0, k0);
  stage_fence();

  for (int c = 0; c < NCHUNK; ++c) {
    const int cur = c & 1;
    if (c + 1 < NCHUNK) copy_stage(cur ^ 1, k0 + (c + 1) * 64);

#pragma unroll
    for (int ks = 0; ks < 64; ks += 32) {
      FragBF a[2], b[2];
      // A fragment (16-bit A 16x32): lanes 0-15 hold K e=0..7 -> K, e=8..15 -> K+16;
      // lanes 16-31 shifted by 8 (ISA 7.12.2)
#pragma unroll
      for (int ms = 0; ms < 2; ++ms) {
        const int m  = wm * 32 + ms * 16 + l;
        const int kb = ks + hi * 8;
        a[ms].q[0] = *(const uint4*)&As[cur][m][kb];
        a[ms].q[1] = *(const uint4*)&As[cur][m][kb + 16];
      }
      // B fragment (32x16): same K order as A: lanes 0-15 e=0..7 -> K, e=8..15 -> K+16; lanes 16-31 shifted by 8
#pragma unroll
      for (int ns = 0; ns < 2; ++ns) {
        const int n  = wn * 32 + ns * 16 + l;
        const int kb = ks + hi * 8;
        b[ns].q[0] = *(const uint4*)&Bs[cur][n][kb];
        b[ns].q[1] = *(const uint4*)&Bs[cur][n][kb + 16];
      }
#pragma unroll
      for (int ms = 0; ms < 2; ++ms)
#pragma unroll
        for (int ns = 0; ns < 2; ++ns)
          acc[ms][ns] = __builtin_amdgcn_wmma_f32_16x16x32_bf16(
              false, a[ms].v, false, b[ns].v, (short)0, acc[ms][ns], false, false);
    }
    stage_fence();
  }

  // C/D layout: VGPR v -> M = v + 8*hi, N = lane&15 (ISA 7.12.2)
  float* op = out_part + (size_t)blockIdx.y * (BATCH * NOUT);
#pragma unroll
  for (int ms = 0; ms < 2; ++ms) {
#pragma unroll
    for (int ns = 0; ns < 2; ++ns) {
      const int mb = wm * 32 + ms * 16 + hi * 8;
      const int nn = n0 + wn * 32 + ns * 16 + l;
#pragma unroll
      for (int v = 0; v < 8; ++v)
        op[(size_t)(mb + v) * NOUT + nn] = acc[ms][ns][v];
    }
  }
}

// ---------------------------------------------------------------------------
// Epilogue 1: sum split-K partials, sigmoid; n<H emits r*h as bf16 (A for G2),
// n>=H emits u (f32).
// ---------------------------------------------------------------------------
__global__ void __launch_bounds__(256) gru_epi1(
    const float* __restrict__ gi_part, const float* __restrict__ Wb,
    const float* __restrict__ h, u16* __restrict__ rhA, float* __restrict__ ubuf)
{
  const int idx = blockIdx.x * 256 + threadIdx.x;
  if (idx >= BATCH * 2 * HID) return;
  const int m = idx >> 11, n = idx & 2047;
  float s = 0.f;
#pragma unroll
  for (int p = 0; p < KSPLIT; ++p) s += gi_part[p * (BATCH * 2 * HID) + idx];
  const float g = 1.f / (1.f + __expf(-(s + Wb[n])));
  if (n < HID) rhA[m * HID + n] = f2bf(g * h[m * HID + n]);
  else         ubuf[m * HID + (n - HID)] = g;
}

// ---------------------------------------------------------------------------
// Epilogue 2: sum partials, tanh, gated update; writes h (f32), h (bf16 for
// next step's A), the output slice, and pre-converts x_{t+1} to bf16.
// ---------------------------------------------------------------------------
__global__ void __launch_bounds__(256) gru_epi2(
    const float* __restrict__ c_part, const float* __restrict__ Ub,
    const float* __restrict__ ubuf, float* __restrict__ h,
    u16* __restrict__ hA, const float* __restrict__ x, u16* __restrict__ xA,
    float* __restrict__ out, int t)
{
  const int idx = blockIdx.x * 256 + threadIdx.x;
  if (idx >= BATCH * HID) return;
  const int m = idx >> 10, n = idx & 1023;
  float s = 0.f;
#pragma unroll
  for (int p = 0; p < KSPLIT; ++p) s += c_part[p * (BATCH * HID) + idx];
  const float cc = tanhf(s + Ub[n]);
  const float u  = ubuf[idx];
  const float hn = u * h[idx] + (1.f - u) * cc;
  h[idx]  = hn;
  hA[idx] = f2bf(hn);
  out[((size_t)m * SEQ + t) * HID + n] = hn;
  if (t + 1 < SEQ)
    xA[idx] = f2bf(x[((size_t)m * SEQ + (t + 1)) * HID + n]);
}

// ---------------------------------------------------------------------------
// One-time setup: cast+transpose W (2048x2048) and U (2048x1024) to bf16,
// stored N-major / K-contiguous so GEMM B-tiles load coalesced. Init h/x bf16.
// ---------------------------------------------------------------------------
__global__ void __launch_bounds__(256) gru_conv_w(const float* __restrict__ W,
                                                  u16* __restrict__ Wt)
{
  const int idx = blockIdx.x * 256 + threadIdx.x;  // over 2048*2048, Wt[n][k]
  if (idx >= 2 * HID * KTOT) return;
  const int n = idx >> 11, k = idx & 2047;
  Wt[idx] = f2bf(W[(size_t)k * (2 * HID) + n]);
}

__global__ void __launch_bounds__(256) gru_conv_u(const float* __restrict__ U,
                                                  u16* __restrict__ Ut)
{
  const int idx = blockIdx.x * 256 + threadIdx.x;  // over 1024*2048, Ut[n][k]
  if (idx >= HID * KTOT) return;
  const int n = idx >> 11, k = idx & 2047;
  Ut[idx] = f2bf(U[(size_t)k * HID + n]);
}

__global__ void __launch_bounds__(256) gru_init(const float* __restrict__ x,
                                                const float* __restrict__ h0,
                                                float* __restrict__ h,
                                                u16* __restrict__ hA,
                                                u16* __restrict__ xA)
{
  const int idx = blockIdx.x * 256 + threadIdx.x;
  if (idx >= BATCH * HID) return;
  const int m = idx >> 10, n = idx & 1023;
  const float hv = h0[idx];
  h[idx]  = hv;
  hA[idx] = f2bf(hv);
  xA[idx] = f2bf(x[(size_t)m * SEQ * DIN + n]);   // x[:, 0, :]
}

// ---------------------------------------------------------------------------
extern "C" void kernel_launch(void* const* d_in, const int* in_sizes, int n_in,
                              void* d_out, int out_size, void* d_ws, size_t ws_size,
                              hipStream_t stream) {
  const float* x  = (const float*)d_in[0];
  const float* h0 = (const float*)d_in[1];
  const float* W  = (const float*)d_in[2];
  const float* Wb = (const float*)d_in[3];
  const float* U  = (const float*)d_in[4];
  const float* Ub = (const float*)d_in[5];
  float* out = (float*)d_out;

  // Workspace carve-up (~19 MB total, all regions 16B aligned)
  char* ws = (char*)d_ws;
  u16* Wt = (u16*)ws;            ws += (size_t)(2 * HID) * KTOT * sizeof(u16); // 8 MB
  u16* Ut = (u16*)ws;            ws += (size_t)HID * KTOT * sizeof(u16);       // 4 MB
  float* gi_part = (float*)ws;   ws += (size_t)KSPLIT * BATCH * 2 * HID * 4;   // 4 MB
  float* c_part  = (float*)ws;   ws += (size_t)KSPLIT * BATCH * HID * 4;       // 2 MB
  float* hbuf    = (float*)ws;   ws += (size_t)BATCH * HID * 4;                // 256 KB
  float* ubuf    = (float*)ws;   ws += (size_t)BATCH * HID * 4;                // 256 KB
  u16* xA  = (u16*)ws;           ws += (size_t)BATCH * DIN * sizeof(u16);      // 128 KB
  u16* hA  = (u16*)ws;           ws += (size_t)BATCH * HID * sizeof(u16);      // 128 KB
  u16* rhA = (u16*)ws;

  // One-time weight cast/transpose + state init (cheap vs. the 512-step scan)
  gru_conv_w<<<(2 * HID * KTOT + 255) / 256, 256, 0, stream>>>(W, Wt);
  gru_conv_u<<<(HID * KTOT + 255) / 256, 256, 0, stream>>>(U, Ut);
  gru_init<<<(BATCH * HID + 255) / 256, 256, 0, stream>>>(x, h0, hbuf, hA, xA);

  const dim3 g1(2 * HID / 128, KSPLIT);  // 16 x 8 = 128 WGs
  const dim3 g2(HID / 128, KSPLIT);      //  8 x 8 =  64 WGs

  for (int t = 0; t < SEQ; ++t) {
    gru_gemm_bf16<2 * HID><<<g1, 256, 0, stream>>>(xA, hA, Wt, gi_part);
    gru_epi1<<<(BATCH * 2 * HID) / 256, 256, 0, stream>>>(gi_part, Wb, hbuf, rhA, ubuf);
    gru_gemm_bf16<HID><<<g2, 256, 0, stream>>>(xA, rhA, Ut, c_part);
    gru_epi2<<<(BATCH * HID) / 256, 256, 0, stream>>>(c_part, Ub, ubuf, hbuf, hA,
                                                      x, xA, out, t);
  }
}
